// AttentionMIL_30915174596859
// MI455X (gfx1250) — matrix operand order repaired and checked by reading, not compiled, not run on hardware
//
#include <hip/hip_runtime.h>
#include <hip/hip_bf16.h>
#include <math.h>

typedef __attribute__((ext_vector_type(16))) __bf16 v16bf;
typedef __attribute__((ext_vector_type(8)))  float  v8f;

#define NB_   16
#define NN_   4096
#define ND_   1024
#define NH_   512
#define NC_   2

union Frag32 {                 // 32-byte WMMA operand assembled from two 16B LDS reads
  struct { uint4 lo, hi; } u;
  v16bf v;
};
union Cvt4 { __bf16 h[4]; uint2 u; };

// -------- fp32 [K,512] -> bf16 TRANSPOSED [512,K] weight conversion (one-time, 2 MB) ------
__global__ __launch_bounds__(256) void cvt_transpose_bf16_kernel(const float* __restrict__ src,
                                                                 __bf16* __restrict__ dst,
                                                                 int K) {
  int idx = blockIdx.x * 256 + threadIdx.x;   // idx over [K,512] row-major
  int k = idx >> 9;
  int c = idx & 511;
  if (k < K) dst[(size_t)c * K + k] = (__bf16)src[idx];
}

// ---------------- fused GEMM: out[rows,512] = epi(A[rows,K] @ W[K,512]) ----------------
// W is supplied pre-transposed as Wt[512,K] bf16 so all staging is 16B-aligned copies.
// EPI 0: LayerNorm(x + bias; over the 512-wide row) * gamma + beta, ReLU -> bf16
// EPI 1: per-row sum( tanh(x + bias) * wa2 ) + ba2 -> fp32 score
// Block: 256 threads (8 wave32), 64 output rows x 512 cols per block.
// Each wave: 4 M-tiles x 4 N-tiles of 16x16, K stepped by 32 (v_wmma_f32_16x16x32_bf16).
template<int K, bool A_BF16, int EPI>
__global__ __launch_bounds__(256) void gemm512_fused_kernel(
    const void* __restrict__ Ain, const __bf16* __restrict__ Wt,
    const float* __restrict__ bias, const float* __restrict__ gamma,
    const float* __restrict__ beta, const float* __restrict__ wa2,
    const float* __restrict__ ba2, __bf16* __restrict__ outbf,
    float* __restrict__ outscore)
{
  constexpr int AS = 40;                    // LDS row strides (x2B = 80B, 16B aligned)
  constexpr int BS = 40;
  __shared__ __bf16 As[64 * AS];            // A tile, row-major  [row][k]   (5.1 KB)
  __shared__ __bf16 Bt[512 * BS];           // B tile, col-major [col][k]   (41 KB)
  __shared__ float  sh_sum[64];
  __shared__ float  sh_sq[64];

  const int tid     = threadIdx.x;
  const int lane    = tid & 31;
  const int wv      = tid >> 5;             // 8 waves
  const int rowBase = blockIdx.x * 64;
  const int colBase = wv * 64;              // wave owns 4 N-tiles
  const int l15     = lane & 15;

  v8f acc[4][4] = {};

  for (int kt = 0; kt < K; kt += 32) {
    // ---- stage A tile 64x32 (convert fp32->bf16 on the fly for layer 1) ----
    {
      const int r = tid >> 2;               // 0..63
      const int k = (tid & 3) * 8;          // 0,8,16,24
      if (A_BF16) {
        const __bf16* Ab = (const __bf16*)Ain + (size_t)(rowBase + r) * K + kt + k;
        *(uint4*)&As[r * AS + k] = *(const uint4*)Ab;
      } else {
        const float* Af = (const float*)Ain + (size_t)(rowBase + r) * K + kt + k;
        float4 v0 = *(const float4*)Af;
        float4 v1 = *(const float4*)(Af + 4);
        Cvt4 c0, c1;
        c0.h[0] = (__bf16)v0.x; c0.h[1] = (__bf16)v0.y; c0.h[2] = (__bf16)v0.z; c0.h[3] = (__bf16)v0.w;
        c1.h[0] = (__bf16)v1.x; c1.h[1] = (__bf16)v1.y; c1.h[2] = (__bf16)v1.z; c1.h[3] = (__bf16)v1.w;
        uint4 o; o.x = c0.u.x; o.y = c0.u.y; o.z = c1.u.x; o.w = c1.u.y;
        *(uint4*)&As[r * AS + k] = o;
      }
    }
    // ---- stage B tile [512 cols][32 k] from pre-transposed Wt: pure 16B copies ----
    #pragma unroll
    for (int i = 0; i < 8; ++i) {
      int c   = tid + i * 256;              // 0..2047 16B chunks
      int col = c >> 2;
      int seg = (c & 3) * 8;
      uint4 v = *(const uint4*)(Wt + (size_t)col * K + kt + seg);
      *(uint4*)&Bt[col * BS + seg] = v;
    }
    if (kt + 32 < K) __builtin_prefetch(Wt + (size_t)(tid) * K + kt + 32, 0, 1);
    __syncthreads();

    // A fragments: lane<16 holds K {0..7,16..23}, lane>=16 holds K {8..15,24..31}, M=lane&15
    Frag32 fa[4];
    const int ak0 = (lane < 16) ? 0 : 8;
    #pragma unroll
    for (int mi = 0; mi < 4; ++mi) {
      const __bf16* p = &As[(mi * 16 + l15) * AS + ak0];
      fa[mi].u.lo = *(const uint4*)p;
      fa[mi].u.hi = *(const uint4*)(p + 16);
    }
    // B fragments: lane<16 holds K {0..7,16..23}, lane>=16 holds K {8..15,24..31}, N = lane&15
    Frag32 fb[4];
    const int bk0 = (lane < 16) ? 0 : 8;
    #pragma unroll
    for (int nj = 0; nj < 4; ++nj) {
      const __bf16* p = &Bt[(colBase + nj * 16 + l15) * BS + bk0];
      fb[nj].u.lo = *(const uint4*)p;
      fb[nj].u.hi = *(const uint4*)(p + 16);
    }
    // 16 back-to-back WMMAs
    #pragma unroll
    for (int nj = 0; nj < 4; ++nj) {
      #pragma unroll
      for (int mi = 0; mi < 4; ++mi) {
        acc[mi][nj] = __builtin_amdgcn_wmma_f32_16x16x32_bf16(
            false, fa[mi].v, false, fb[nj].v, (short)0, acc[mi][nj], false, false);
      }
    }
    __syncthreads();
  }

  // ---- fused epilogue. C layout: lane<16 -> rows mi*16+0..7, lane>=16 -> rows mi*16+8..15,
  //      col = colBase + nj*16 + (lane&15), VGPR r -> row offset r. ----
  float bv[4];
  #pragma unroll
  for (int nj = 0; nj < 4; ++nj) bv[nj] = bias[colBase + nj * 16 + l15];

  if (tid < 64) { sh_sum[tid] = 0.f; sh_sq[tid] = 0.f; }
  __syncthreads();

  const int rhalf = (lane >> 4) << 3;       // 0 or 8

  if constexpr (EPI == 0) {
    #pragma unroll
    for (int mi = 0; mi < 4; ++mi) {
      #pragma unroll
      for (int r = 0; r < 8; ++r) {
        float s = 0.f, q = 0.f;
        #pragma unroll
        for (int nj = 0; nj < 4; ++nj) {
          float v = acc[mi][nj][r] + bv[nj];
          acc[mi][nj][r] = v;
          s += v; q += v * v;
        }
        #pragma unroll
        for (int m = 8; m >= 1; m >>= 1) {  // reduce the 16-lane column group in-register
          s += __shfl_xor(s, m, 32);
          q += __shfl_xor(q, m, 32);
        }
        if (l15 == 0) {                     // one LDS atomic per (wave, row)
          int row = mi * 16 + rhalf + r;
          atomicAdd(&sh_sum[row], s);
          atomicAdd(&sh_sq[row], q);
        }
      }
    }
    __syncthreads();
    float gv[4], be[4];
    #pragma unroll
    for (int nj = 0; nj < 4; ++nj) {
      int col = colBase + nj * 16 + l15;
      gv[nj] = gamma[col]; be[nj] = beta[col];
    }
    #pragma unroll
    for (int mi = 0; mi < 4; ++mi) {
      #pragma unroll
      for (int r = 0; r < 8; ++r) {
        int row    = mi * 16 + rhalf + r;
        float mean = sh_sum[row] * (1.f / 512.f);
        float var  = sh_sq[row] * (1.f / 512.f) - mean * mean;
        float rstd = rsqrtf(var + 1e-5f);
        #pragma unroll
        for (int nj = 0; nj < 4; ++nj) {
          int col = colBase + nj * 16 + l15;
          float v = (acc[mi][nj][r] - mean) * rstd * gv[nj] + be[nj];
          v = fmaxf(v, 0.f);
          outbf[(size_t)(rowBase + row) * 512 + col] = (__bf16)v;
        }
      }
    }
  } else {
    float wv2[4];
    #pragma unroll
    for (int nj = 0; nj < 4; ++nj) wv2[nj] = wa2[colBase + nj * 16 + l15];
    #pragma unroll
    for (int mi = 0; mi < 4; ++mi) {
      #pragma unroll
      for (int r = 0; r < 8; ++r) {
        float s = 0.f;
        #pragma unroll
        for (int nj = 0; nj < 4; ++nj)
          s += tanhf(acc[mi][nj][r] + bv[nj]) * wv2[nj];
        #pragma unroll
        for (int m = 8; m >= 1; m >>= 1) s += __shfl_xor(s, m, 32);
        if (l15 == 0) atomicAdd(&sh_sum[mi * 16 + rhalf + r], s);
      }
    }
    __syncthreads();
    if (tid < 64) outscore[rowBase + tid] = sh_sum[tid] + ba2[0];
  }
}

// ---------------- per-bag softmax stats: max + sum(exp) over N=4096 ----------------
__global__ __launch_bounds__(256) void bag_stats_kernel(const float* __restrict__ scores,
                                                        float* __restrict__ bagst) {
  __shared__ float red[256];
  const int b = blockIdx.x;
  const float* s = scores + (size_t)b * NN_;
  float m = -1e30f;
  for (int n = threadIdx.x; n < NN_; n += 256) m = fmaxf(m, s[n]);
  red[threadIdx.x] = m;
  __syncthreads();
  for (int st = 128; st > 0; st >>= 1) {
    if (threadIdx.x < st) red[threadIdx.x] = fmaxf(red[threadIdx.x], red[threadIdx.x + st]);
    __syncthreads();
  }
  m = red[0];
  __syncthreads();
  float sum = 0.f;
  for (int n = threadIdx.x; n < NN_; n += 256) sum += __expf(s[n] - m);
  red[threadIdx.x] = sum;
  __syncthreads();
  for (int st = 128; st > 0; st >>= 1) {
    if (threadIdx.x < st) red[threadIdx.x] += red[threadIdx.x + st];
    __syncthreads();
  }
  if (threadIdx.x == 0) { bagst[b * 2] = m; bagst[b * 2 + 1] = red[0]; }
}

// ---------------- pooled[b,h] = sum_n softmax(scores)[b,n] * h2[b,n,h] ----------------
__global__ __launch_bounds__(256) void pool_kernel(const float* __restrict__ scores,
                                                   const float* __restrict__ bagst,
                                                   const __bf16* __restrict__ h2,
                                                   float* __restrict__ pooled) {
  __shared__ float red[256];
  const int b     = blockIdx.x >> 3;
  const int chunk = blockIdx.x & 7;
  const int col   = chunk * 64 + (threadIdx.x & 63);
  const int sub   = threadIdx.x >> 6;       // 0..3
  const float m   = bagst[b * 2];
  const float inv = 1.f / bagst[b * 2 + 1];
  const float* s  = scores + (size_t)b * NN_;
  const __bf16* h = h2 + (size_t)b * NN_ * NH_;
  float acc = 0.f;
  for (int n = sub; n < NN_; n += 4)
    acc += __expf(s[n] - m) * inv * (float)h[(size_t)n * NH_ + col];
  red[threadIdx.x] = acc;
  __syncthreads();
  if (sub == 0) {
    acc += red[64 + threadIdx.x] + red[128 + threadIdx.x] + red[192 + threadIdx.x];
    pooled[(size_t)b * NH_ + col] = acc;
  }
}

// ---------------- tiny classifier: logits = relu(pooled@Wc1+bc1)@Wc2+bc2 ----------------
__global__ __launch_bounds__(256) void classifier_kernel(const float* __restrict__ pooled,
                                                         const float* __restrict__ Wc1,
                                                         const float* __restrict__ bc1,
                                                         const float* __restrict__ Wc2,
                                                         const float* __restrict__ bc2,
                                                         float* __restrict__ out) {
  __shared__ float ps[NB_ * NH_];
  __shared__ float t1[NB_ * NH_];
  for (int i = threadIdx.x; i < NB_ * NH_; i += 256) ps[i] = pooled[i];
  __syncthreads();
  for (int o = threadIdx.x; o < NB_ * NH_; o += 256) {
    int row = o >> 9, col = o & 511;
    float s = bc1[col];
    for (int k = 0; k < NH_; ++k) s = fmaf(ps[row * NH_ + k], Wc1[(size_t)k * NH_ + col], s);
    t1[o] = fmaxf(s, 0.f);
  }
  __syncthreads();
  if (threadIdx.x < NB_ * NC_) {
    int row = threadIdx.x >> 1, c = threadIdx.x & 1;
    float s = bc2[c];
    for (int k = 0; k < NH_; ++k) s = fmaf(t1[row * NH_ + k], Wc2[k * NC_ + c], s);
    out[row * NC_ + c] = s;
  }
}

extern "C" void kernel_launch(void* const* d_in, const int* in_sizes, int n_in,
                              void* d_out, int out_size, void* d_ws, size_t ws_size,
                              hipStream_t stream) {
  (void)in_sizes; (void)n_in; (void)out_size; (void)ws_size;
  const float* x   = (const float*)d_in[0];
  const float* W1  = (const float*)d_in[1];
  const float* b1  = (const float*)d_in[2];
  const float* g1  = (const float*)d_in[3];
  const float* be1 = (const float*)d_in[4];
  const float* W2  = (const float*)d_in[5];
  const float* b2  = (const float*)d_in[6];
  const float* g2  = (const float*)d_in[7];
  const float* be2 = (const float*)d_in[8];
  const float* Wa1 = (const float*)d_in[9];
  const float* ba1 = (const float*)d_in[10];
  const float* wa2 = (const float*)d_in[11];
  const float* ba2 = (const float*)d_in[12];
  const float* Wc1 = (const float*)d_in[13];
  const float* bc1 = (const float*)d_in[14];
  const float* Wc2 = (const float*)d_in[15];
  const float* bc2 = (const float*)d_in[16];
  float* out = (float*)d_out;

  char* ws = (char*)d_ws;
  size_t off = 0;
  auto carve = [&](size_t bytes) -> void* {
    void* p = ws + off;
    off += (bytes + 255) & ~(size_t)255;
    return p;
  };
  __bf16* W1t    = (__bf16*)carve((size_t)ND_ * NH_ * 2);   // transposed [512,1024]
  __bf16* W2t    = (__bf16*)carve((size_t)NH_ * NH_ * 2);   // transposed [512,512]
  __bf16* Wa1t   = (__bf16*)carve((size_t)NH_ * NH_ * 2);   // transposed [512,512]
  __bf16* h1     = (__bf16*)carve((size_t)NB_ * NN_ * NH_ * 2);
  __bf16* h2     = (__bf16*)carve((size_t)NB_ * NN_ * NH_ * 2);
  float*  scores = (float*)carve((size_t)NB_ * NN_ * 4);
  float*  bagst  = (float*)carve((size_t)NB_ * 2 * 4);
  float*  pooled = (float*)carve((size_t)NB_ * NH_ * 4);

  // one-time weight convert + transpose to bf16 [col][k]
  cvt_transpose_bf16_kernel<<<(ND_ * NH_) / 256, 256, 0, stream>>>(W1, W1t, ND_);
  cvt_transpose_bf16_kernel<<<(NH_ * NH_) / 256, 256, 0, stream>>>(W2, W2t, NH_);
  cvt_transpose_bf16_kernel<<<(NH_ * NH_) / 256, 256, 0, stream>>>(Wa1, Wa1t, NH_);

  const int rows = NB_ * NN_;   // 65536, divisible by 64

  // layer 1: x @ W1 -> LN -> ReLU  (K=1024, A in fp32)
  gemm512_fused_kernel<ND_, false, 0><<<rows / 64, 256, 0, stream>>>(
      x, W1t, b1, g1, be1, nullptr, nullptr, h1, nullptr);
  // layer 2: h1 @ W2 -> LN -> ReLU (K=512, A in bf16)
  gemm512_fused_kernel<NH_, true, 0><<<rows / 64, 256, 0, stream>>>(
      h1, W2t, b2, g2, be2, nullptr, nullptr, h2, nullptr);
  // attention scores: sum(tanh(h2 @ Wa1 + ba1) * wa2) + ba2
  gemm512_fused_kernel<NH_, true, 1><<<rows / 64, 256, 0, stream>>>(
      h2, Wa1t, ba1, nullptr, nullptr, wa2, ba2, nullptr, scores);

  bag_stats_kernel<<<NB_, 256, 0, stream>>>(scores, bagst);
  pool_kernel<<<NB_ * 8, 256, 0, stream>>>(scores, bagst, h2, pooled);
  classifier_kernel<<<1, 256, 0, stream>>>(pooled, Wc1, bc1, Wc2, bc2, out);
}
